// Point_Recongeneration_for_context1_6047313953105
// MI455X (gfx1250) — matrix operand order repaired and checked by reading, not compiled, not run on hardware
//
#include <hip/hip_runtime.h>
#include <hip/hip_bf16.h>

// ---------------------------------------------------------------------------
// Types for WMMA operands (CDNA5 wave32: bf16 A/B 16x32 & 32x16, f32 C/D 8 VGPR)
// ---------------------------------------------------------------------------
typedef __attribute__((ext_vector_type(16))) __bf16 v16bf;
typedef __attribute__((ext_vector_type(8)))  __bf16 v8bf;
typedef __attribute__((ext_vector_type(8)))  float  v8f;
typedef __attribute__((ext_vector_type(4)))  unsigned int v4u;
typedef __attribute__((ext_vector_type(8)))  int v8i;
typedef __attribute__((ext_vector_type(4)))  int v4i;

union ABFrag { v16bf v; v8bf h8[2]; };

__device__ __forceinline__ v8bf zero_v8bf() {
  v8bf z;
  #pragma unroll
  for (int i = 0; i < 8; ++i) z[i] = (__bf16)0.0f;
  return z;
}

#define SCONV_HAVE_TDM __has_builtin(__builtin_amdgcn_tensor_load_to_lds)

// Stage `bytes` (multiple of 8, bytes/8 < 65536) from global `src` into LDS
// `dst` using the Tensor Data Mover (async DMA, TENSORcnt-tracked).
// Issued by wave 0 only; caller must __syncthreads() afterwards.
__device__ __forceinline__ void tdm_stage_weights(const __bf16* src, __bf16* dst,
                                                  unsigned bytes) {
#if SCONV_HAVE_TDM
  if ((threadIdx.x >> 5) == 0) {
    const unsigned long long ga = (unsigned long long)(const void*)src;
    const unsigned lds_addr = (unsigned)(unsigned long long)(void*)dst;
    const unsigned n8 = bytes >> 3;  // 8-byte elements (data_size=3)

    // D# group0: count=1 | lds_addr | global_addr[56:0] | type=2
    v4u g0 = {1u,
              lds_addr,
              (unsigned)(ga & 0xffffffffu),
              (unsigned)((ga >> 32) & 0x01ffffffu) | (2u << 30)};
    // D# group1: wg_mask=0, data_size=3 (8B); tensor_dim0=tile_dim0=n8;
    // tensor_dim1=tile_dim1=1; tile_dim2=0; stride0=n8; stride1=0.
    v8i g1 = {(int)(3u << 16),                                   // [17:16] data_size
              (int)((n8 & 0xffffu) << 16),                        // tensor_dim0[15:0]
              (int)(((n8 >> 16) & 0xffffu) | (1u << 16)),         // dim0[31:16], dim1=1
              (int)((n8 & 0xffffu) << 16),                        // dim1 hi=0, tile_dim0
              1,                                                  // tile_dim1=1, tile_dim2=0
              (int)n8,                                            // stride0[31:0]
              0, 0};
    v4i gz = {0, 0, 0, 0};
#if __clang_major__ >= 23
    v8i gz8 = {0, 0, 0, 0, 0, 0, 0, 0};
    __builtin_amdgcn_tensor_load_to_lds(g0, g1, gz, gz, gz8, 0);
#else
    __builtin_amdgcn_tensor_load_to_lds(g0, g1, gz, gz, 0);
#endif
    __builtin_amdgcn_s_wait_tensorcnt(0);
  }
#else
  for (unsigned i = threadIdx.x * 8u; i < (bytes >> 1); i += blockDim.x * 8u)
    *(v8bf*)(dst + i) = *(const v8bf*)(src + i);
#endif
}

// ---------------------------------------------------------------------------
// Sparse conv: out[p, :] (+= resid) = act( sum_k Fp[nb[p,k]] @ W[k] + bias )
//   Fin : [NP, CIN] bf16 (row NP-1 is the zero sentinel row)
//   Wt  : [27, COUT, CIN] bf16 (pre-transposed so B fragments are contiguous)
// One wave = 16 points x full COUT. Weights staged in dynamic LDS per WG (TDM).
// ---------------------------------------------------------------------------
template<int CIN, int COUT, int RELU, int WAVES>
__global__ __launch_bounds__(WAVES * 32)
void sconv_wmma(const __bf16* __restrict__ Fin,
                const int*    __restrict__ nb,
                const __bf16* __restrict__ Wt,
                const float*  __restrict__ bias,   // may be null, len COUT
                const __bf16* __restrict__ resid,  // may be null, stride ostride
                __bf16* __restrict__ out,          // stride ostride
                int ostride, int n)
{
  constexpr int KT = 27;
  constexpr int WELEMS = KT * COUT * CIN;
  static_assert((WELEMS * 2) % 8 == 0 && (WELEMS * 2 / 8) < 65536, "TDM tile");
  extern __shared__ __align__(16) char smem_raw[];
  __bf16* wlds = (__bf16*)smem_raw;

  tdm_stage_weights(Wt, wlds, (unsigned)(WELEMS * 2));
  __syncthreads();

  const int lane = threadIdx.x & 31;
  const int wave = threadIdx.x >> 5;
  const int m    = lane & 15;   // point row within tile / output column
  const int h    = lane >> 4;   // half-wave selector
  const int pbase = (blockIdx.x * WAVES + wave) * 16;
  if (pbase >= n) return;

  constexpr int NT = COUT / 16;
  v8f acc[NT];
  #pragma unroll
  for (int t = 0; t < NT; ++t) {
    #pragma unroll
    for (int j = 0; j < 8; ++j) acc[t][j] = 0.0f;
  }

  int pm = pbase + m;
  if (pm >= n) pm = n - 1;
  const int* nbrow = nb + (size_t)pm * KT;

  #pragma unroll 1
  for (int k = 0; k < KT; ++k) {
    const int idx = nbrow[k];                        // sentinel n -> zero row
    const __bf16* row = Fin + (size_t)idx * CIN;
    if (k + 1 < KT)                                   // global_prefetch_b8
      __builtin_prefetch(Fin + (size_t)nbrow[k + 1] * CIN, 0, 0);

    if constexpr (CIN >= 32) {
      #pragma unroll
      for (int ch = 0; ch < CIN / 32; ++ch) {
        ABFrag a;  // A 16x32 bf16: lane half h holds K = 16h..16h+15
        a.h8[0] = *(const v8bf*)(row + ch * 32 +     16 * h);
        a.h8[1] = *(const v8bf*)(row + ch * 32 + 8 + 16 * h);
        #pragma unroll
        for (int ct = 0; ct < NT; ++ct) {
          ABFrag b;  // B 32x16 bf16: lane half h holds K = 16h..16h+15, n = m
          const __bf16* wp =
              wlds + (size_t)(k * COUT + ct * 16 + m) * CIN + ch * 32 + 16 * h;
          b.h8[0] = *(const v8bf*)(wp);
          b.h8[1] = *(const v8bf*)(wp + 8);
          acc[ct] = __builtin_amdgcn_wmma_f32_16x16x32_bf16(
              false, a.v, false, b.v, (short)0, acc[ct], false, false);
        }
      }
    } else {  // CIN == 16: zero-pad K dimension to 32
      ABFrag a;
      a.h8[0] = h == 0 ? *(const v8bf*)(row) : zero_v8bf();
      a.h8[1] = h == 0 ? *(const v8bf*)(row + 8) : zero_v8bf();
      #pragma unroll
      for (int ct = 0; ct < NT; ++ct) {
        ABFrag b;
        if (h == 0) {
          const __bf16* wp = wlds + (size_t)(k * COUT + ct * 16 + m) * CIN;
          b.h8[0] = *(const v8bf*)(wp);
          b.h8[1] = *(const v8bf*)(wp + 8);
        } else {
          b.h8[0] = zero_v8bf();
          b.h8[1] = zero_v8bf();
        }
        acc[ct] = __builtin_amdgcn_wmma_f32_16x16x32_bf16(
            false, a.v, false, b.v, (short)0, acc[ct], false, false);
      }
    }
  }

  // Epilogue: D layout -> lane (m=col, h) VGPR r => point pbase + r + 8h
  #pragma unroll
  for (int ct = 0; ct < NT; ++ct) {
    #pragma unroll
    for (int r = 0; r < 8; ++r) {
      const int prow = pbase + r + 8 * h;
      const int col  = ct * 16 + m;
      if (prow < n) {
        float v = acc[ct][r];
        if (bias)  v += bias[col];
        if (RELU)  v = v > 0.0f ? v : 0.0f;
        if (resid) v += (float)resid[(size_t)prow * ostride + col];
        out[(size_t)prow * ostride + col] = (__bf16)v;
      }
    }
  }
}

// ---------------------------------------------------------------------------
// Dense 1x1: out[p,:] (+= resid) = act( Fin[p,:] @ Wt^T )   Wt: [COUT][CIN]
// ---------------------------------------------------------------------------
template<int CIN, int COUT, int RELU, int WAVES>
__global__ __launch_bounds__(WAVES * 32)
void gemm_wmma(const __bf16* __restrict__ Fin,
               const __bf16* __restrict__ Wt,
               const __bf16* __restrict__ resid,
               __bf16* __restrict__ out,
               int ostride, int n)
{
  const int lane = threadIdx.x & 31;
  const int wave = threadIdx.x >> 5;
  const int m    = lane & 15;
  const int h    = lane >> 4;
  const int pbase = (blockIdx.x * WAVES + wave) * 16;
  if (pbase >= n) return;

  constexpr int NT = COUT / 16;
  v8f acc[NT];
  #pragma unroll
  for (int t = 0; t < NT; ++t) {
    #pragma unroll
    for (int j = 0; j < 8; ++j) acc[t][j] = 0.0f;
  }

  int pm = pbase + m;
  if (pm >= n) pm = n - 1;
  const __bf16* row = Fin + (size_t)pm * CIN;

  if constexpr (CIN >= 32) {
    #pragma unroll
    for (int ch = 0; ch < CIN / 32; ++ch) {
      ABFrag a;
      a.h8[0] = *(const v8bf*)(row + ch * 32 +     16 * h);
      a.h8[1] = *(const v8bf*)(row + ch * 32 + 8 + 16 * h);
      #pragma unroll
      for (int ct = 0; ct < NT; ++ct) {
        ABFrag b;
        const __bf16* wp = Wt + (size_t)(ct * 16 + m) * CIN + ch * 32 + 16 * h;
        b.h8[0] = *(const v8bf*)(wp);
        b.h8[1] = *(const v8bf*)(wp + 8);
        acc[ct] = __builtin_amdgcn_wmma_f32_16x16x32_bf16(
            false, a.v, false, b.v, (short)0, acc[ct], false, false);
      }
    }
  } else {  // CIN == 16
    ABFrag a;
    a.h8[0] = h == 0 ? *(const v8bf*)(row) : zero_v8bf();
    a.h8[1] = h == 0 ? *(const v8bf*)(row + 8) : zero_v8bf();
    #pragma unroll
    for (int ct = 0; ct < NT; ++ct) {
      ABFrag b;
      if (h == 0) {
        const __bf16* wp = Wt + (size_t)(ct * 16 + m) * CIN;
        b.h8[0] = *(const v8bf*)(wp);
        b.h8[1] = *(const v8bf*)(wp + 8);
      } else {
        b.h8[0] = zero_v8bf();
        b.h8[1] = zero_v8bf();
      }
      acc[ct] = __builtin_amdgcn_wmma_f32_16x16x32_bf16(
          false, a.v, false, b.v, (short)0, acc[ct], false, false);
    }
  }

  #pragma unroll
  for (int ct = 0; ct < NT; ++ct) {
    #pragma unroll
    for (int r = 0; r < 8; ++r) {
      const int prow = pbase + r + 8 * h;
      const int col  = ct * 16 + m;
      if (prow < n) {
        float v = acc[ct][r];
        if (RELU)  v = v > 0.0f ? v : 0.0f;
        if (resid) v += (float)resid[(size_t)prow * ostride + col];
        out[(size_t)prow * ostride + col] = (__bf16)v;
      }
    }
  }
}

// ---------------------------------------------------------------------------
// Helpers: casts / transposes / sentinels / classifier / prune
// ---------------------------------------------------------------------------
__global__ void cast_pad_kernel(const float* __restrict__ x, __bf16* __restrict__ o,
                                size_t valid, size_t total) {
  size_t i = (size_t)blockIdx.x * blockDim.x + threadIdx.x;
  if (i >= total) return;
  o[i] = (i < valid) ? (__bf16)x[i] : (__bf16)0.0f;
}

// W[k][ci][co] (f32) -> o[k][co][ci] (bf16)
__global__ void tcast_w_kernel(const float* __restrict__ W, __bf16* __restrict__ o,
                               int K, int Cin, int Cout) {
  int i = blockIdx.x * blockDim.x + threadIdx.x;
  int total = K * Cin * Cout;
  if (i >= total) return;
  int co = i % Cout;
  int t  = i / Cout;
  int ci = t % Cin;
  int k  = t / Cin;
  o[((size_t)k * Cout + co) * Cin + ci] = (__bf16)W[i];
}

__global__ void zero_sentinel_kernel(__bf16* fb, __bf16* u, __bf16* v) {
  int t = threadIdx.x;
  if (t < 64) fb[t] = (__bf16)0.0f;
  if (t < 16) { u[t] = (__bf16)0.0f; v[t] = (__bf16)0.0f; }
}

__global__ void cls_kernel(const __bf16* __restrict__ F, const int* __restrict__ nb,
                           const float* __restrict__ Wc, const float* __restrict__ bcls,
                           float* __restrict__ cls, int n) {
  __shared__ float wl[27 * 64];
  for (int i = threadIdx.x; i < 27 * 64; i += blockDim.x) wl[i] = Wc[i];
  __syncthreads();
  int pnt = blockIdx.x * blockDim.x + threadIdx.x;
  if (pnt >= n) return;
  float acc = bcls[0];
  const int* nbr = nb + (size_t)pnt * 27;
  for (int k = 0; k < 27; ++k) {
    const __bf16* row = F + (size_t)nbr[k] * 64;
    const float*  w   = wl + k * 64;
    #pragma unroll
    for (int c = 0; c < 64; ++c) acc += (float)row[c] * w[c];
  }
  cls[pnt] = acc;
}

__global__ void prune_kernel(const __bf16* __restrict__ F, const float* __restrict__ cls,
                             const int* __restrict__ target, float* __restrict__ outF,
                             float* __restrict__ keep, int n) {
  int pnt = blockIdx.x * blockDim.x + threadIdx.x;
  if (pnt >= n) return;
  bool kp = (cls[pnt] > 0.0f) || (target[pnt] != 0);
  float s = kp ? 1.0f : 0.0f;
  const __bf16* row = F + (size_t)pnt * 64;
  float* orow = outF + (size_t)pnt * 64;
  #pragma unroll
  for (int c = 0; c < 64; ++c) orow[c] = s * (float)row[c];
  keep[pnt] = s;
}

// ---------------------------------------------------------------------------
// Host orchestration
// ---------------------------------------------------------------------------
extern "C" void kernel_launch(void* const* d_in, const int* in_sizes, int n_in,
                              void* d_out, int out_size, void* d_ws, size_t ws_size,
                              hipStream_t stream) {
  (void)n_in; (void)out_size;
  const int n  = in_sizes[0] / 64;
  const int NP = n + 1;

  const float* x      = (const float*)d_in[0];
  const int*   nb     = (const int*)d_in[1];
  const int*   target = (const int*)d_in[2];
  const float* W_conv = (const float*)d_in[3];
  const float* b_conv = (const float*)d_in[4];
  const float* w00    = (const float*)d_in[5];   // [2,27,64,16]
  const float* w01    = (const float*)d_in[6];   // [2,27,16,32]
  const float* w10    = (const float*)d_in[7];   // [2,64,16]
  const float* w11    = (const float*)d_in[8];   // [2,27,16,16]
  const float* w12    = (const float*)d_in[9];   // [2,16,32]
  const float* W_cls  = (const float*)d_in[10];  // [27,64,1]
  const float* b_cls  = (const float*)d_in[11];

  // workspace carve-up (256B aligned regions)
  char* p = (char*)d_ws;
  auto alloc = [&](size_t bytes) -> char* {
    char* r = p; p += (bytes + 255) & ~(size_t)255; return r;
  };
  __bf16* FA   = (__bf16*)alloc((size_t)NP * 64 * 2);
  __bf16* FB   = (__bf16*)alloc((size_t)NP * 64 * 2);
  __bf16* U    = (__bf16*)alloc((size_t)NP * 16 * 2);
  __bf16* V    = (__bf16*)alloc((size_t)NP * 16 * 2);
  __bf16* Wc1  = (__bf16*)alloc((size_t)27 * 64 * 64 * 2);
  __bf16* W00t = (__bf16*)alloc((size_t)2 * 27 * 16 * 64 * 2);
  __bf16* W01t = (__bf16*)alloc((size_t)2 * 27 * 32 * 16 * 2);
  __bf16* W10t = (__bf16*)alloc((size_t)2 * 16 * 64 * 2);
  __bf16* W11t = (__bf16*)alloc((size_t)2 * 27 * 16 * 16 * 2);
  __bf16* W12t = (__bf16*)alloc((size_t)2 * 32 * 16 * 2);
  if ((size_t)(p - (char*)d_ws) > ws_size) return;

  // input cast (+ zero sentinel row) and weight transpose/casts
  {
    size_t total = (size_t)NP * 64, valid = (size_t)n * 64;
    cast_pad_kernel<<<(unsigned)((total + 255) / 256), 256, 0, stream>>>(x, FA, valid, total);
  }
  auto tc = [&](const float* w, __bf16* o, int K, int Ci, int Co) {
    int tot = K * Ci * Co;
    tcast_w_kernel<<<(tot + 255) / 256, 256, 0, stream>>>(w, o, K, Ci, Co);
  };
  tc(W_conv, Wc1, 27, 64, 64);
  for (int i = 0; i < 2; ++i) {
    tc(w00 + (size_t)i * 27 * 64 * 16, W00t + (size_t)i * 27 * 16 * 64, 27, 64, 16);
    tc(w01 + (size_t)i * 27 * 16 * 32, W01t + (size_t)i * 27 * 32 * 16, 27, 16, 32);
    tc(w10 + (size_t)i * 64 * 16,      W10t + (size_t)i * 16 * 64,       1, 64, 16);
    tc(w11 + (size_t)i * 27 * 16 * 16, W11t + (size_t)i * 27 * 16 * 16, 27, 16, 16);
    tc(w12 + (size_t)i * 16 * 32,      W12t + (size_t)i * 32 * 16,       1, 16, 32);
  }
  zero_sentinel_kernel<<<1, 64, 0, stream>>>(FB + (size_t)n * 64,
                                             U + (size_t)n * 16,
                                             V + (size_t)n * 16);

  constexpr int WAVES = 8;
  const int tiles = (n + 15) / 16;
  const int grid  = (tiles + WAVES - 1) / WAVES;
  dim3 blk(WAVES * 32);

  // Layer 1: relu(sconv64->64 + b_conv)   FA -> FB
  sconv_wmma<64, 64, 1, WAVES><<<grid, blk, 27 * 64 * 64 * 2, stream>>>(
      FA, nb, Wc1, b_conv, nullptr, FB, 64, n);

  // Two inception-residual blocks, ping-pong FB <-> FA
  __bf16* F = FB;
  __bf16* G = FA;
  for (int i = 0; i < 2; ++i) {
    const __bf16* ww00 = W00t + (size_t)i * 27 * 16 * 64;
    const __bf16* ww01 = W01t + (size_t)i * 27 * 32 * 16;
    const __bf16* ww10 = W10t + (size_t)i * 16 * 64;
    const __bf16* ww11 = W11t + (size_t)i * 27 * 16 * 16;
    const __bf16* ww12 = W12t + (size_t)i * 32 * 16;

    // branch1: u = relu(F @ w10); v = relu(sconv16(u, w11))
    gemm_wmma<64, 16, 1, WAVES><<<grid, blk, 0, stream>>>(F, ww10, nullptr, U, 16, n);
    sconv_wmma<16, 16, 1, WAVES><<<grid, blk, 27 * 16 * 16 * 2, stream>>>(
        U, nb, ww11, nullptr, nullptr, V, 16, n);
    // branch0: h = relu(sconv(F, w00)) (reuse U); out0 = sconv(h, w01) -> G[:,0:32] + F[:,0:32]
    sconv_wmma<64, 16, 1, WAVES><<<grid, blk, 27 * 16 * 64 * 2, stream>>>(
        F, nb, ww00, nullptr, nullptr, U, 16, n);
    sconv_wmma<16, 32, 0, WAVES><<<grid, blk, 27 * 32 * 16 * 2, stream>>>(
        U, nb, ww01, nullptr, F, G, 64, n);
    // branch1 tail: out1 = v @ w12 -> G[:,32:64] + F[:,32:64]
    gemm_wmma<16, 32, 0, WAVES><<<grid, blk, 0, stream>>>(V, ww12, F + 32, G + 32, 64, n);

    __bf16* t = F; F = G; G = t;   // after i=1: F == FB holds final features
  }

  float* outF = (float*)d_out;
  float* cls  = outF + (size_t)n * 64;
  float* keep = cls + n;

  cls_kernel<<<(n + 255) / 256, 256, 0, stream>>>(F, nb, W_cls, b_cls, cls, n);
  prune_kernel<<<(n + 255) / 256, 256, 0, stream>>>(F, cls, target, outF, keep, n);
}
